// ParallelDroplessMLP_2302102471512
// MI455X (gfx1250) — matrix operand order repaired and checked by reading, not compiled, not run on hardware
//
#include <hip/hip_runtime.h>
#include <hip/hip_bf16.h>

// ---------- problem constants (from reference setup) ----------
#define HS        512      // hidden size (GEMM1 K, GEMM2 N)
#define FFN       2048     // ffn size   (GEMM1 N, GEMM2 K)
#define NEXP      8
#define TOPK      2
#define MT        64       // slot-tile rows per block
#define NC        128      // ffn chunk width
#define MAX_TILES 144      // >= 8192/64 + 8
#define MAX_SLOTS 8704     // >= 8192 + 8*63 (padded)

// padded LDS strides (bf16 elements)
#define HSP  520           // xg row stride
#define HCP  136           // h-chunk row stride
#define KP   40            // weight-panel k stride (per n row)

// ws int layout (routing region)
#define WS_COUNTS   0
#define WS_CURSORS  8
#define WS_PADOFF   16
#define WS_NTILES   24
#define WS_TILE_E   32
#define WS_TILE_R   (32 + MAX_TILES)
#define WS_SLOT_TOK (32 + 2 * MAX_TILES)
#define WS_SLOT_W   (WS_SLOT_TOK + MAX_SLOTS)          // float region
#define WS_INT_END  (WS_SLOT_W + MAX_SLOTS)
#define WS_BF16_OFF ((WS_INT_END + 63) & ~63)          // 256B-aligned int offset for bf16 weights

typedef __attribute__((ext_vector_type(16))) __bf16 v16bf;
typedef __attribute__((ext_vector_type(8)))  __bf16 v8bf;
typedef __attribute__((ext_vector_type(8)))  float  v8f;
typedef int gv4i __attribute__((vector_size(16)));     // 128-bit payload type for async builtin

union V16 { v16bf v; v8bf h[2]; };

// ---- CDNA5 async global->LDS copy (ASYNCcnt), guarded for toolchain support ----
#if defined(__has_builtin)
#if __has_builtin(__builtin_amdgcn_global_load_async_to_lds_b128) && \
    __has_builtin(__builtin_amdgcn_s_wait_asynccnt)
#define USE_ASYNC_LDS 1
#endif
#endif

__device__ __forceinline__ void copy16(const __bf16* g, __bf16* l) {
#ifdef USE_ASYNC_LDS
    __builtin_amdgcn_global_load_async_to_lds_b128((gv4i*)g, (gv4i*)l, 0, 0);
#else
    *reinterpret_cast<uint4*>(l) = *reinterpret_cast<const uint4*>(g);
#endif
}
__device__ __forceinline__ void wait_async_done() {
#ifdef USE_ASYNC_LDS
    __builtin_amdgcn_s_wait_asynccnt(0);
#endif
}

// A-operand (16x32 bf16): lane holds row m=lane&15; K-halves selected by lane>>4.
__device__ __forceinline__ v16bf ldA(const __bf16* p) {
    V16 u;
    u.h[0] = *(const v8bf*)(p);
    u.h[1] = *(const v8bf*)(p + 16);
    return u.v;
}
// B-operand (32x16 bf16) from transposed panel [n][k]: lane holds col n=lane&15,
// K-halves selected by lane>>4 (k = 8h+s, 16+8h+s-8).
__device__ __forceinline__ v16bf ldB(const __bf16* p) {
    V16 u;
    u.h[0] = *(const v8bf*)(p);
    u.h[1] = *(const v8bf*)(p + 16);
    return u.v;
}

__device__ __forceinline__ float gelu_exact(float v) {
    return 0.5f * v * (1.0f + erff(v * 0.70710678118654752f));
}

// ---------------- routing kernels ----------------
__global__ void k_hist(const int* __restrict__ idx, int* __restrict__ wsI, int nA) {
    int t = blockIdx.x * blockDim.x + threadIdx.x;
    if (t < nA) atomicAdd(&wsI[WS_COUNTS + idx[t]], 1);
}

__global__ void k_plan(int* __restrict__ wsI) {
    for (int i = threadIdx.x; i < MAX_SLOTS; i += blockDim.x)
        wsI[WS_SLOT_TOK + i] = -1;
    if (threadIdx.x == 0) {
        int po = 0, t = 0;
        for (int e = 0; e < NEXP; ++e) {
            wsI[WS_PADOFF + e]  = po;
            wsI[WS_CURSORS + e] = 0;
            int c   = wsI[WS_COUNTS + e];
            int ntl = (c + MT - 1) / MT;
            for (int r = 0; r < ntl; ++r) {
                wsI[WS_TILE_E + t] = e;
                wsI[WS_TILE_R + t] = po + r * MT;
                ++t;
            }
            po += ntl * MT;
        }
        wsI[WS_NTILES] = t;
    }
}

__global__ void k_scatter(const int* __restrict__ idx, const float* __restrict__ wts,
                          int* __restrict__ wsI, float* __restrict__ slot_w, int nA) {
    int a = blockIdx.x * blockDim.x + threadIdx.x;
    if (a < nA) {
        int e   = idx[a];
        int pos = wsI[WS_PADOFF + e] + atomicAdd(&wsI[WS_CURSORS + e], 1);
        wsI[WS_SLOT_TOK + pos] = a / TOPK;
        slot_w[pos]            = wts[a];
    }
}

// ---------------- one-time weight transpose+convert (f32 -> bf16, [e][n][k]) ----------------
__global__ void k_cvt_w1(const float* __restrict__ in, __bf16* __restrict__ outT) {
    size_t base = (size_t)(blockIdx.x * blockDim.x + threadIdx.x);
    size_t gsz  = (size_t)gridDim.x * blockDim.x;
#pragma unroll
    for (int j = 0; j < 4; ++j) {
        size_t i = base + (size_t)j * gsz;                // i over [e][k(HS)][n(FFN)]
        int e = (int)(i >> 20), r = (int)(i & 0xFFFFF);
        int k = r >> 11, n = r & (FFN - 1);
        outT[((size_t)e << 20) + (size_t)n * HS + k] = (__bf16)in[i];
    }
}
__global__ void k_cvt_w2(const float* __restrict__ in, __bf16* __restrict__ outT) {
    size_t base = (size_t)(blockIdx.x * blockDim.x + threadIdx.x);
    size_t gsz  = (size_t)gridDim.x * blockDim.x;
#pragma unroll
    for (int j = 0; j < 4; ++j) {
        size_t i = base + (size_t)j * gsz;                // i over [e][k(FFN)][n(HS)]
        int e = (int)(i >> 20), r = (int)(i & 0xFFFFF);
        int k = r >> 9, n = r & (HS - 1);
        outT[((size_t)e << 20) + (size_t)n * FFN + k] = (__bf16)in[i];
    }
}

// ---------------- panel staging (double-buffered) ----------------
__device__ __forceinline__ void stage_pan1(int tid, __bf16* buf, const float* W1,
                                           const __bf16* W1T, int useT, int fc, int ks) {
    if (useT) {
        for (int c = tid; c < 4 * NC; c += 256) {          // 16B chunks
            int n = c >> 2, j = c & 3;
            copy16(W1T + (size_t)(fc + n) * HS + ks + j * 8, buf + n * KP + j * 8);
        }
    } else {
        for (int i = tid; i < 32 * NC; i += 256) {
            int n = i & (NC - 1), k = i >> 7;
            buf[n * KP + k] = (__bf16)W1[(size_t)(ks + k) * FFN + fc + n];
        }
    }
}
__device__ __forceinline__ void stage_pan2(int tid, __bf16* buf, const float* W2,
                                           const __bf16* W2T, int useT, int kf) {
    if (useT) {
        for (int c = tid; c < 4 * HS; c += 256) {          // 16B chunks
            int n = c >> 2, j = c & 3;
            copy16(W2T + (size_t)n * FFN + kf + j * 8, buf + n * KP + j * 8);
        }
    } else {
        for (int i = tid; i < 32 * HS; i += 256) {
            int n = i & (HS - 1), k = i >> 9;
            buf[n * KP + k] = (__bf16)W2[(size_t)(kf + k) * HS + n];
        }
    }
}

// ---------------- fused grouped MLP (bf16 WMMA) ----------------
__global__ __launch_bounds__(256, 1)
void moe_fused(const float* __restrict__ x,
               const float* __restrict__ w1,
               const float* __restrict__ w2,
               const __bf16* __restrict__ w1t,
               const __bf16* __restrict__ w2t,
               int useT,
               const int*   __restrict__ meta,
               const int*   __restrict__ tile_e,
               const int*   __restrict__ tile_r,
               const int*   __restrict__ slot_tok,
               const float* __restrict__ slot_w,
               float*       __restrict__ out) {
    if ((int)blockIdx.x >= meta[0]) return;

    extern __shared__ __bf16 lds[];
    __bf16* xg   = lds;                                   // [MT][HSP]
    __bf16* hbuf = lds + MT * HSP;                        // [MT][HCP]
    __bf16* panbuf[2] = { lds + MT * HSP + MT * HCP,      // [512][KP] x2 (double buffer)
                          lds + MT * HSP + MT * HCP + HS * KP };

    const int e    = tile_e[blockIdx.x];
    const int row0 = tile_r[blockIdx.x];
    const float*  W1  = w1  + (size_t)e * HS * FFN;       // [HS][FFN]   f32
    const float*  W2  = w2  + (size_t)e * FFN * HS;       // [FFN][HS]   f32
    const __bf16* W1T = w1t + ((size_t)e << 20);          // [FFN][HS]   bf16 (n-major)
    const __bf16* W2T = w2t + ((size_t)e << 20);          // [HS][FFN]   bf16 (n-major)

    const int tid  = threadIdx.x;
    const int wave = tid >> 5;
    const int lane = tid & 31;
    const int half = lane >> 4;
    const int l15  = lane & 15;
    const int wm   = wave & 1;
    const int wn   = wave >> 1;

    // gather + convert activation tile: [MT][HS] f32 -> bf16 in LDS
    for (int i = tid; i < MT * HS; i += 256) {
        int r = i >> 9, c = i & (HS - 1);
        int tok = slot_tok[row0 + r];
        float v = (tok >= 0) ? x[(size_t)tok * HS + c] : 0.0f;
        xg[r * HSP + c] = (__bf16)v;
    }
    __syncthreads();

    v8f acc2[2][8];
#pragma unroll
    for (int mi = 0; mi < 2; ++mi)
#pragma unroll
        for (int ni = 0; ni < 8; ++ni)
            acc2[mi][ni] = (v8f){0.f,0.f,0.f,0.f,0.f,0.f,0.f,0.f};

    for (int fc = 0; fc < FFN; fc += NC) {
        // ---- GEMM1: h_chunk[64][NC] = xg @ W1[:, fc:fc+NC] ----
        v8f acc1[2][2];
#pragma unroll
        for (int mi = 0; mi < 2; ++mi)
#pragma unroll
            for (int ni = 0; ni < 2; ++ni)
                acc1[mi][ni] = (v8f){0.f,0.f,0.f,0.f,0.f,0.f,0.f,0.f};

        stage_pan1(tid, panbuf[0], W1, W1T, useT, fc, 0);
        for (int ksi = 0; ksi < HS / 32; ++ksi) {
            wait_async_done();
            __syncthreads();
            if (ksi + 1 < HS / 32)
                stage_pan1(tid, panbuf[(ksi + 1) & 1], W1, W1T, useT, fc, (ksi + 1) * 32);
            const __bf16* pb = panbuf[ksi & 1];
            const int ks = ksi * 32;
#pragma unroll
            for (int mi = 0; mi < 2; ++mi) {
                v16bf a = ldA(xg + ((2 * wm + mi) * 16 + l15) * HSP + ks + 8 * half);
#pragma unroll
                for (int ni = 0; ni < 2; ++ni) {
                    v16bf b = ldB(pb + (wn * 32 + ni * 16 + l15) * KP + 8 * half);
                    acc1[mi][ni] = __builtin_amdgcn_wmma_f32_16x16x32_bf16(
                        false, a, false, b, (short)0, acc1[mi][ni], false, false);
                }
            }
        }

        // GELU (exact) + store h-chunk as bf16 (D layout: m = r + 8*half, n = l15)
#pragma unroll
        for (int mi = 0; mi < 2; ++mi)
#pragma unroll
            for (int ni = 0; ni < 2; ++ni)
#pragma unroll
                for (int r = 0; r < 8; ++r) {
                    float v = gelu_exact(acc1[mi][ni][r]);
                    int m = (2 * wm + mi) * 16 + 8 * half + r;
                    int n = wn * 32 + ni * 16 + l15;
                    hbuf[m * HCP + n] = (__bf16)v;
                }

        // ---- GEMM2: acc2 += h_chunk @ W2[fc:fc+NC, :] ----
        stage_pan2(tid, panbuf[0], W2, W2T, useT, fc);
        for (int ksi = 0; ksi < NC / 32; ++ksi) {
            wait_async_done();
            __syncthreads();
            if (ksi + 1 < NC / 32)
                stage_pan2(tid, panbuf[(ksi + 1) & 1], W2, W2T, useT, fc + (ksi + 1) * 32);
            const __bf16* pb = panbuf[ksi & 1];
            const int ks = ksi * 32;
#pragma unroll
            for (int mi = 0; mi < 2; ++mi) {
                v16bf a = ldA(hbuf + ((2 * wm + mi) * 16 + l15) * HCP + ks + 8 * half);
#pragma unroll
                for (int ni = 0; ni < 8; ++ni) {
                    v16bf b = ldB(pb + (wn * 128 + ni * 16 + l15) * KP + 8 * half);
                    acc2[mi][ni] = __builtin_amdgcn_wmma_f32_16x16x32_bf16(
                        false, a, false, b, (short)0, acc2[mi][ni], false, false);
                }
            }
        }
    }

    // ---- scatter-combine: out[token] += weight * y ----
#pragma unroll
    for (int mi = 0; mi < 2; ++mi) {
        int mbase = (2 * wm + mi) * 16 + 8 * half;
#pragma unroll
        for (int r = 0; r < 8; ++r) {
            int slot = row0 + mbase + r;
            int tok  = slot_tok[slot];
            if (tok >= 0) {
                float wgt = slot_w[slot];
#pragma unroll
                for (int ni = 0; ni < 8; ++ni) {
                    int n = wn * 128 + ni * 16 + l15;
                    atomicAdd(out + (size_t)tok * HS + n, wgt * acc2[mi][ni][r]);
                }
            }
        }
    }
}

// ---------------- launcher ----------------
extern "C" void kernel_launch(void* const* d_in, const int* in_sizes, int n_in,
                              void* d_out, int out_size, void* d_ws, size_t ws_size,
                              hipStream_t stream) {
    const float* x  = (const float*)d_in[0];
    const float* ew = (const float*)d_in[1];
    const int*   ei = (const int*)d_in[2];
    const float* w1 = (const float*)d_in[3];
    const float* w2 = (const float*)d_in[4];
    float* out = (float*)d_out;
    int*   wsI = (int*)d_ws;
    float* wsF = (float*)d_ws;

    const int T  = in_sizes[0] / HS;  // 4096 tokens
    const int nA = T * TOPK;          // 8192 assignments

    (void)hipMemsetAsync(d_out, 0, (size_t)out_size * sizeof(float), stream);
    (void)hipMemsetAsync(d_ws, 0, 16 * sizeof(int), stream);

    k_hist<<<(nA + 255) / 256, 256, 0, stream>>>(ei, wsI, nA);
    k_plan<<<1, 256, 0, stream>>>(wsI);
    k_scatter<<<(nA + 255) / 256, 256, 0, stream>>>(ei, ew, wsI, wsF + WS_SLOT_W, nA);

    // bf16 pre-transposed weight cache in ws (if it fits): [e][n][k], L2-resident (32 MiB)
    const size_t elemsW = (size_t)NEXP * HS * FFN;                 // 2^23 per tensor
    __bf16* w1t = (__bf16*)(wsI + WS_BF16_OFF);
    __bf16* w2t = w1t + elemsW;
    const size_t need = (size_t)WS_BF16_OFF * sizeof(int) + 2 * elemsW * sizeof(__bf16);
    const int useT = (ws_size >= need) ? 1 : 0;
    if (useT) {
        const int cvtBlocks = (int)(elemsW / (256 * 4));           // 8192
        k_cvt_w1<<<cvtBlocks, 256, 0, stream>>>(w1, w1t);
        k_cvt_w2<<<cvtBlocks, 256, 0, stream>>>(w2, w2t);
    }

    const size_t ldsBytes = (size_t)(MT * HSP + MT * HCP + 2 * HS * KP) * sizeof(__bf16); // ~162 KB
    moe_fused<<<MAX_TILES, 256, ldsBytes, stream>>>(
        x, w1, w2, w1t, w2t, useT,
        wsI + WS_NTILES, wsI + WS_TILE_E, wsI + WS_TILE_R,
        wsI + WS_SLOT_TOK, wsF + WS_SLOT_W, out);
}
